// SG_21208548508411
// MI455X (gfx1250) — matrix operand order repaired and checked by reading, not compiled, not run on hardware
//
#include <hip/hip_runtime.h>
#include <stdint.h>

// ---------------------------------------------------------------------------
// Problem constants (match reference: B=16, N=4096, D=128, S=512, K=32, C=256)
// ---------------------------------------------------------------------------
#define B_ 16
#define N_ 4096
#define D_ 128
#define S_ 512
#define K_ 32
#define C_ 256
#define M_ (B_ * S_ * K_)          // 262144 GEMM rows
#define GEMM_BLOCKS (M_ / 128)     // 2048
#define EPS_ 1e-5f

typedef __attribute__((ext_vector_type(16))) __bf16 v16bf;
typedef __attribute__((ext_vector_type(8)))  float  v8f;
typedef __attribute__((ext_vector_type(4)))  unsigned int v4u;

union FragU { v16bf bf; v4u u[2]; };

__device__ __forceinline__ unsigned short bf16b(float f) {
  unsigned int u = __float_as_uint(f);
  u += 0x7fffu + ((u >> 16) & 1u);           // round-to-nearest-even
  return (unsigned short)(u >> 16);
}
__device__ __forceinline__ float b2f(unsigned short h) {
  return __uint_as_float(((unsigned int)h) << 16);
}

// ---------------------------------------------------------------------------
// 1) Transpose x [B,D,N] -> xt [B,N,D]  (LDS tiled, coalesced both sides)
// ---------------------------------------------------------------------------
__global__ __launch_bounds__(256) void sg_transpose(const float* __restrict__ x,
                                                    float* __restrict__ xt) {
  __shared__ float tile[32][33];
  const int b  = blockIdx.z;
  const int n0 = blockIdx.x * 32;
  const int d0 = blockIdx.y * 32;
  const int tx = threadIdx.x & 31;
  const int ty = threadIdx.x >> 5;           // 0..7
#pragma unroll
  for (int i = 0; i < 4; i++) {
    int dl = ty + i * 8;
    tile[dl][tx] = x[((size_t)b * D_ + (d0 + dl)) * N_ + n0 + tx];
  }
  __syncthreads();
#pragma unroll
  for (int i = 0; i < 4; i++) {
    int nl = ty + i * 8;
    xt[((size_t)b * N_ + (n0 + nl)) * D_ + d0 + tx] = tile[tx][nl];
  }
}

// ---------------------------------------------------------------------------
// 2) Farthest point sampling, deterministic start at 0. One block per batch.
// ---------------------------------------------------------------------------
__global__ __launch_bounds__(256) void sg_fps(const float* __restrict__ coords,
                                              int* __restrict__ fpsidx) {
  __shared__ float dists[N_];
  __shared__ float rv[256];
  __shared__ int   ri[256];
  const int b = blockIdx.x, tid = threadIdx.x;
  const float* cb = coords + (size_t)b * N_ * 3;
  for (int i = tid; i < N_; i += 256) dists[i] = 1.0e10f;
  __syncthreads();
  int far = 0;
  for (int t = 0; t < S_; t++) {
    if (tid == 0) fpsidx[b * S_ + t] = far;
    const float cx = cb[far * 3 + 0], cy = cb[far * 3 + 1], cz = cb[far * 3 + 2];
    float bv = -1.0f; int bi = 0;
    for (int i = tid; i < N_; i += 256) {
      float dx = cb[i * 3 + 0] - cx, dy = cb[i * 3 + 1] - cy, dz = cb[i * 3 + 2] - cz;
      float d = dx * dx + dy * dy + dz * dz;
      float nd = fminf(dists[i], d);
      dists[i] = nd;
      if (nd > bv) { bv = nd; bi = i; }        // ascending i -> first max kept
    }
    rv[tid] = bv; ri[tid] = bi;
    __syncthreads();
    for (int off = 128; off > 0; off >>= 1) {
      if (tid < off) {
        float ov = rv[tid + off]; int oi = ri[tid + off];
        if (ov > rv[tid] || (ov == rv[tid] && oi < ri[tid])) { rv[tid] = ov; ri[tid] = oi; }
      }
      __syncthreads();
    }
    far = ri[0];
    __syncthreads();
  }
}

// ---------------------------------------------------------------------------
// 3) Gather: new_xyz -> d_out[0:B*S*3], newfeat[bs][0:128] = xt[b][fps[bs]][:]
// ---------------------------------------------------------------------------
__global__ __launch_bounds__(128) void sg_gather(const float* __restrict__ coords,
                                                 const float* __restrict__ xt,
                                                 const int* __restrict__ fpsidx,
                                                 float* __restrict__ newfeat,
                                                 float* __restrict__ out_xyz) {
  const int bs = blockIdx.x, c = threadIdx.x;
  const int b = bs >> 9;
  const int idx = fpsidx[bs];
  newfeat[(size_t)bs * D_ + c] = xt[((size_t)b * N_ + idx) * D_ + c];
  if (c < 3) out_xyz[(size_t)bs * 3 + c] = coords[((size_t)b * N_ + idx) * 3 + c];
}

// ---------------------------------------------------------------------------
// 4) KNN: per (b,s) compute 4096 squared distances, extract 32 smallest
//    (iterative min with smallest-index tie-break -> matches top_k set).
// ---------------------------------------------------------------------------
__global__ __launch_bounds__(256) void sg_knn(const float* __restrict__ coords,
                                              const int* __restrict__ fpsidx,
                                              int* __restrict__ knn) {
  __shared__ float dsq[N_];
  __shared__ float rv[256];
  __shared__ int   ri[256];
  const int bs = blockIdx.x, tid = threadIdx.x;
  const int b = bs >> 9;
  const float* cb = coords + (size_t)b * N_ * 3;
  const int cidx = fpsidx[bs];
  const float cx = cb[cidx * 3 + 0], cy = cb[cidx * 3 + 1], cz = cb[cidx * 3 + 2];
  for (int i = tid; i < N_; i += 256) {
    float dx = cb[i * 3 + 0] - cx, dy = cb[i * 3 + 1] - cy, dz = cb[i * 3 + 2] - cz;
    dsq[i] = dx * dx + dy * dy + dz * dz;
  }
  __syncthreads();
  for (int k = 0; k < K_; k++) {
    float bv = 3.0e38f; int bi = 0;
    for (int i = tid; i < N_; i += 256) {
      float v = dsq[i];
      if (v < bv) { bv = v; bi = i; }
    }
    rv[tid] = bv; ri[tid] = bi;
    __syncthreads();
    for (int off = 128; off > 0; off >>= 1) {
      if (tid < off) {
        float ov = rv[tid + off]; int oi = ri[tid + off];
        if (ov < rv[tid] || (ov == rv[tid] && oi < ri[tid])) { rv[tid] = ov; ri[tid] = oi; }
      }
      __syncthreads();
    }
    if (tid == 0) { int w = ri[0]; knn[(size_t)bs * K_ + k] = w; dsq[w] = 3.0e38f; }
    __syncthreads();
  }
}

// ---------------------------------------------------------------------------
// 5) Pre-pack weights into per-lane WMMA B fragments (bf16).
//    pb[nt][ks][lane][j] u32 ; lane<16: n=nt*16+lane, K=ks*32+2j{,+1}
//                              lane>=16: same n, K=ks*32+16+2j{,+1}
//    B[k][n] = W[n][k]  (GEMM: out[m][n] = sum_k A[m][k]*W[n][k])
// ---------------------------------------------------------------------------
__global__ __launch_bounds__(256) void sg_prep_weights(const float* __restrict__ w1,
                                                       const float* __restrict__ w2,
                                                       unsigned int* __restrict__ pb1,
                                                       unsigned int* __restrict__ pb2) {
  const int g = blockIdx.x * 256 + threadIdx.x;      // 0..65535
  const int mat = g >> 15;
  const int e = g & 32767;
  const int j = e & 7;
  const int lane = (e >> 3) & 31;
  const int ks = (e >> 8) & 7;
  const int nt = (e >> 11) & 15;
  const int n = nt * 16 + (lane & 15);
  const int k0 = ks * 32 + ((lane >> 4) << 4) + (j << 1);
  const float* w = mat ? w2 : w1;
  unsigned int lo = bf16b(w[(size_t)n * 256 + k0]);
  unsigned int hi = bf16b(w[(size_t)n * 256 + k0 + 1]);
  unsigned int* dst = mat ? pb2 : pb1;
  dst[e] = lo | (hi << 16);
}

// ---------------------------------------------------------------------------
// 6) WMMA GEMM: [M x 256] x [256 x 256] in bf16, fp32 accumulate.
//    MODE 0: A row m=(bs,k) = [ xt[b][knn] - newfeat | newfeat ]  (conv1)
//    MODE 1: A row m = relu(bn1_a * Y1[m] + bn1_b)               (conv2)
//    Epilogue: +bias, store bf16 Y, exact fp32 per-channel sum/sumsq partials.
// ---------------------------------------------------------------------------
union SmemGemm {
  unsigned short a[128 * 256];   // 64 KB A tile (bf16), row stride 512 B
  float red[8192];               // 32 KB stat slots: [n][16] sums, +4096 sumsq
};

template <int MODE>
__global__ __launch_bounds__(256) void sg_gemm(const float* __restrict__ xt,
                                               const int* __restrict__ knn,
                                               const float* __restrict__ newfeat,
                                               const unsigned short* __restrict__ Yin,
                                               const float* __restrict__ bnab,
                                               const unsigned int* __restrict__ pb,
                                               const float* __restrict__ bias,
                                               unsigned short* __restrict__ Yout,
                                               float* __restrict__ gsum,
                                               float* __restrict__ gsq) {
  __shared__ SmemGemm sm;
  const int tid = threadIdx.x;
  const int m0 = blockIdx.x * 128;

  // ---- stage A tile (128 rows x 256 cols, bf16) ----
  if constexpr (MODE == 0) {
    const int r = tid >> 1, half = tid & 1, c0 = half * 64;
    const int m = m0 + r, bs = m >> 5, k = m & 31, b = bs >> 9;
    const int idx = knn[(size_t)bs * K_ + k];
    const float* src = xt + ((size_t)b * N_ + idx) * D_;
    const float* nfr = newfeat + (size_t)bs * D_;
    unsigned short* arow = sm.a + r * 256;
#pragma unroll 8
    for (int j = 0; j < 64; j++) {
      int c = c0 + j;
      float nf = nfr[c], g = src[c];
      arow[c] = bf16b(g - nf);       // recentred neighbor features
      arow[128 + c] = bf16b(nf);     // broadcast centroid features
    }
  } else {
    const int r = tid >> 1, half = tid & 1, c0 = half * 128;
    const unsigned short* src = Yin + (size_t)(m0 + r) * 256;
    unsigned short* arow = sm.a + r * 256;
#pragma unroll 8
    for (int j = 0; j < 128; j++) {
      int c = c0 + j;
      float v = fmaf(bnab[c], b2f(src[c]), bnab[256 + c]);
      arow[c] = bf16b(fmaxf(v, 0.0f));
    }
  }
  __syncthreads();

  // ---- WMMA main loop: wave owns 16 rows x 256 cols ----
  const int lane = tid & 31, wave = tid >> 5;
  const int hi = lane >> 4;                       // half-wave selector
  v8f zero;
#pragma unroll
  for (int j = 0; j < 8; j++) zero[j] = 0.0f;
  v8f acc[16];
#pragma unroll
  for (int i = 0; i < 16; i++) acc[i] = zero;

  const int rowb = wave * 16 + (lane & 15);
  const char* abase = (const char*)sm.a + rowb * 512;

#pragma unroll
  for (int ks = 0; ks < 8; ks++) {
    if (ks < 7)  // pull next K-step of weight fragments toward L0/L2
      __builtin_prefetch(pb + ((((size_t)0 * 8 + (ks + 1)) * 32 + lane) << 3), 0, 0);
    FragU af;
    af.u[0] = *(const v4u*)(abase + ks * 64 + hi * 32);        // K 0-7 / 16-23
    af.u[1] = *(const v4u*)(abase + ks * 64 + 16 + hi * 32);   // K 8-15 / 24-31
#pragma unroll
    for (int nt = 0; nt < 16; nt++) {
      FragU bf_;
      const v4u* bp = (const v4u*)(pb + ((((size_t)nt * 8 + ks) * 32 + lane) << 3));
      bf_.u[0] = bp[0];
      bf_.u[1] = bp[1];
      acc[nt] = __builtin_amdgcn_wmma_f32_16x16x32_bf16(
          false, af.bf, false, bf_.bf, (short)0, acc[nt], false, false);
    }
  }

  // ---- epilogue: bias add, bf16 store, exact fp32 BN partial stats ----
  __syncthreads();   // done with sm.a, reuse as sm.red
#pragma unroll
  for (int nt = 0; nt < 16; nt++) {
    const int n = nt * 16 + (lane & 15);
    const float bi_ = bias[n];
    const size_t mbase = (size_t)m0 + wave * 16 + hi * 8;
    float lsum = 0.0f, lsq = 0.0f;
#pragma unroll
    for (int r2 = 0; r2 < 8; r2++) {
      float v = acc[nt][r2] + bi_;
      Yout[(mbase + r2) * 256 + n] = bf16b(v);
      lsum += v;
      lsq += v * v;
    }
    sm.red[n * 16 + wave * 2 + hi] = lsum;
    sm.red[4096 + n * 16 + wave * 2 + hi] = lsq;
  }
  __syncthreads();
  {
    float s = 0.0f, q = 0.0f;
#pragma unroll
    for (int i = 0; i < 16; i++) {            // fixed order -> deterministic
      s += sm.red[tid * 16 + i];
      q += sm.red[4096 + tid * 16 + i];
    }
    gsum[(size_t)blockIdx.x * 256 + tid] = s;
    gsq[(size_t)blockIdx.x * 256 + tid] = q;
  }
}

// ---------------------------------------------------------------------------
// 7) Finish BN stats: one block per channel; a = g/sqrt(var+eps), b = beta-mean*a
// ---------------------------------------------------------------------------
__global__ __launch_bounds__(256) void sg_reduce_stats(const float* __restrict__ gsum,
                                                       const float* __restrict__ gsq,
                                                       const float* __restrict__ gamma,
                                                       const float* __restrict__ beta,
                                                       float* __restrict__ ab,
                                                       float invM) {
  __shared__ float ss[256], qq[256];
  const int c = blockIdx.x, tid = threadIdx.x;
  float s = 0.0f, q = 0.0f;
  for (int i = tid; i < GEMM_BLOCKS; i += 256) {
    s += gsum[(size_t)i * 256 + c];
    q += gsq[(size_t)i * 256 + c];
  }
  ss[tid] = s; qq[tid] = q;
  __syncthreads();
  for (int off = 128; off > 0; off >>= 1) {
    if (tid < off) { ss[tid] += ss[tid + off]; qq[tid] += qq[tid + off]; }
    __syncthreads();
  }
  if (tid == 0) {
    float mean = ss[0] * invM;
    float var = qq[0] * invM - mean * mean;
    float a = gamma[c] * rsqrtf(var + EPS_);
    ab[c] = a;
    ab[256 + c] = beta[c] - mean * a;
  }
}

// ---------------------------------------------------------------------------
// 8) BN2 affine + ReLU + max over K -> out[b][c][s]
// ---------------------------------------------------------------------------
__global__ __launch_bounds__(256) void sg_final(const unsigned short* __restrict__ Y2,
                                                const float* __restrict__ ab,
                                                float* __restrict__ out) {
  const int bs = blockIdx.x, c = threadIdx.x;
  const int b = bs >> 9, s = bs & 511;
  const float a = ab[c], bb = ab[256 + c];
  const size_t mbase = (size_t)bs * K_;
  float best = 0.0f;                         // relu(x) >= 0, so 0 is the identity
#pragma unroll 4
  for (int k = 0; k < K_; k++) {
    float v = fmaf(a, b2f(Y2[(mbase + k) * 256 + c]), bb);
    best = fmaxf(best, fmaxf(v, 0.0f));
  }
  out[((size_t)b * C_ + c) * S_ + s] = best;
}

// ---------------------------------------------------------------------------
// Host launcher
// ---------------------------------------------------------------------------
extern "C" void kernel_launch(void* const* d_in, const int* in_sizes, int n_in,
                              void* d_out, int out_size, void* d_ws, size_t ws_size,
                              hipStream_t stream) {
  const float* x       = (const float*)d_in[0];
  const float* coords  = (const float*)d_in[1];
  const float* conv1_w = (const float*)d_in[2];
  const float* conv1_b = (const float*)d_in[3];
  const float* bn1_g   = (const float*)d_in[4];
  const float* bn1_bt  = (const float*)d_in[5];
  const float* conv2_w = (const float*)d_in[6];
  const float* conv2_b = (const float*)d_in[7];
  const float* bn2_g   = (const float*)d_in[8];
  const float* bn2_bt  = (const float*)d_in[9];
  float* out = (float*)d_out;

  char* ws = (char*)d_ws;
  size_t off = 0;
  auto take = [&](size_t bytes) -> char* {
    char* p = ws + off;
    off = (off + bytes + 255) & ~(size_t)255;
    return p;
  };
  float*          xt      = (float*)take((size_t)B_ * N_ * D_ * 4);      // 32 MB
  int*            fpsidx  = (int*)take((size_t)B_ * S_ * 4);
  int*            knn     = (int*)take((size_t)B_ * S_ * K_ * 4);
  float*          newfeat = (float*)take((size_t)B_ * S_ * D_ * 4);
  unsigned short* Y1      = (unsigned short*)take((size_t)M_ * C_ * 2);  // 134 MB
  unsigned short* Y2      = (unsigned short*)take((size_t)M_ * C_ * 2);  // 134 MB
  unsigned int*   pb1     = (unsigned int*)take(32768 * 4);
  unsigned int*   pb2     = (unsigned int*)take(32768 * 4);
  float*          gsum1   = (float*)take((size_t)GEMM_BLOCKS * C_ * 4);
  float*          gsq1    = (float*)take((size_t)GEMM_BLOCKS * C_ * 4);
  float*          gsum2   = (float*)take((size_t)GEMM_BLOCKS * C_ * 4);
  float*          gsq2    = (float*)take((size_t)GEMM_BLOCKS * C_ * 4);
  float*          bn1ab   = (float*)take(512 * 4);
  float*          bn2ab   = (float*)take(512 * 4);

  const float invM = 1.0f / (float)M_;

  sg_prep_weights<<<256, 256, 0, stream>>>(conv1_w, conv2_w, pb1, pb2);
  sg_transpose<<<dim3(N_ / 32, D_ / 32, B_), 256, 0, stream>>>(x, xt);
  sg_fps<<<B_, 256, 0, stream>>>(coords, fpsidx);
  sg_gather<<<B_ * S_, 128, 0, stream>>>(coords, xt, fpsidx, newfeat, out);
  sg_knn<<<B_ * S_, 256, 0, stream>>>(coords, fpsidx, knn);

  sg_gemm<0><<<GEMM_BLOCKS, 256, 0, stream>>>(xt, knn, newfeat, nullptr, nullptr,
                                              pb1, conv1_b, Y1, gsum1, gsq1);
  sg_reduce_stats<<<C_, 256, 0, stream>>>(gsum1, gsq1, bn1_g, bn1_bt, bn1ab, invM);
  sg_gemm<1><<<GEMM_BLOCKS, 256, 0, stream>>>(nullptr, nullptr, nullptr, Y1, bn1ab,
                                              pb2, conv2_b, Y2, gsum2, gsq2);
  sg_reduce_stats<<<C_, 256, 0, stream>>>(gsum2, gsq2, bn2_g, bn2_bt, bn2ab, invM);

  sg_final<<<B_ * S_, 256, 0, stream>>>(Y2, bn2ab, out + (size_t)B_ * S_ * 3);
}
